// PoseAwareAttention_47115791237846
// MI455X (gfx1250) — matrix operand order repaired and checked by reading, not compiled, not run on hardware
//
#include <hip/hip_runtime.h>
#include <hip/hip_bf16.h>

// ---------------------------------------------------------------------------
// PoseAwareAttention on MI455X (gfx1250): bf16 WMMA pipeline, f32 accumulate,
// async-to-LDS triple-buffered GEMM mainloops (1 barrier / K-step).
// ---------------------------------------------------------------------------

#define DIM   1024
#define HEADS 16
#define HDIM  64
#define BSZ   8
#define NSEQ  1024
#define MTOT  (BSZ * NSEQ)   // 8192 rows

typedef __attribute__((ext_vector_type(16))) __bf16  v16bf;
typedef __attribute__((ext_vector_type(8)))  float   v8f;
typedef __attribute__((ext_vector_type(4)))  unsigned int u32x4;

union Frag { v16bf v; u32x4 q[2]; };

__device__ __forceinline__ unsigned short f32_to_bf16(float f) {
    unsigned int u = __builtin_bit_cast(unsigned int, f);
    u += 0x7FFFu + ((u >> 16) & 1u);           // round-to-nearest-even
    return (unsigned short)(u >> 16);
}

__device__ __forceinline__ v8f wmma_bf16(v16bf a, v16bf b, v8f c) {
    // D = A(16x32 bf16) * B(32x16 bf16) + C(16x16 f32)
    return __builtin_amdgcn_wmma_f32_16x16x32_bf16(
        false, a, false, b, (short)0, c, false, false);
}

// ---------------------------------------------------------------------------
// elementwise f32 -> bf16
__global__ __launch_bounds__(256) void cvt_f32_bf16(const float* __restrict__ in,
                                                    unsigned short* __restrict__ out,
                                                    int n) {
    int i = blockIdx.x * 256 + threadIdx.x;
    if (i < n) out[i] = f32_to_bf16(in[i]);
}

// transpose + convert: in f32 [K][Nc] row-major  ->  out bf16 [Nc][K]
__global__ __launch_bounds__(256) void transpose_cvt(const float* __restrict__ in,
                                                     unsigned short* __restrict__ out,
                                                     int K, int Nc) {
    __shared__ float tile[32][33];
    int n0 = blockIdx.x * 32, k0 = blockIdx.y * 32;
    int tx = threadIdx.x, ty = threadIdx.y;
    #pragma unroll
    for (int i = ty; i < 32; i += 8)
        tile[i][tx] = in[(size_t)(k0 + i) * Nc + n0 + tx];
    __syncthreads();
    #pragma unroll
    for (int i = ty; i < 32; i += 8)
        out[(size_t)(n0 + i) * K + k0 + tx] = f32_to_bf16(tile[tx][i]);
}

// pose MLP: relu(pose @ W1 + b1) @ W2 + b2 -> pose_emb [8][3072]
__global__ __launch_bounds__(256) void pose_mlp(const float* __restrict__ pose,
                                                const float* __restrict__ W1,
                                                const float* __restrict__ b1,
                                                const float* __restrict__ W2,
                                                const float* __restrict__ b2,
                                                float* __restrict__ pose_emb) {
    __shared__ float h[64];
    int b = blockIdx.y;
    int t = threadIdx.x;
    if (t < 64) {
        float a = b1[t];
        #pragma unroll
        for (int i = 0; i < 3; ++i) a += pose[b * 3 + i] * W1[i * 64 + t];
        h[t] = fmaxf(a, 0.0f);
    }
    __syncthreads();
    int c = blockIdx.x * 256 + t;
    float acc = b2[c];
    #pragma unroll 8
    for (int i = 0; i < 64; ++i) acc += h[i] * W2[(size_t)i * (3 * DIM) + c];
    pose_emb[(size_t)b * (3 * DIM) + c] = acc;
}

// ---------------------------------------------------------------------------
// Block-tiled 128x128 GEMM mainloop, K-step 32, TRIPLE-buffered LDS fed by
// gfx1250 async-to-LDS copies. A: [M][K] row-major, Bt: [Ncol][K] row-major.
// Buffer p occupies bytes [p*16384, p*16384+16384): A tile then B tile (8KB ea).
// Stage issued at iter i writes buf (i+1)%3; buffers being read are (i)%3 and
// (laggards) (i+2)%3 -> no WAR conflict -> single barrier per K-step.
// smem is the kernel's single __shared__ allocation => LDS base offset 0.
__device__ __forceinline__ void gemm_block_128x128(const unsigned short* __restrict__ A,
                                                   const unsigned short* __restrict__ Bt,
                                                   int K, int m0, int n0,
                                                   unsigned short* smem,
                                                   v8f acc[2][4]) {
    const int tid  = threadIdx.x;
    const int lane = tid & 31, wave = tid >> 5;
    const int g = lane >> 4, ln = lane & 15;
    const int wm = (wave & 3) * 32;          // wave row offset inside block tile
    const int wn = (wave >> 2) * 64;         // wave col offset inside block tile

    // Capture the LDS pointer into an asm so the compiler must assume the
    // memory-clobbering async copies below can write smem (otherwise the
    // fragment loads get folded to undef and deleted).
    asm volatile("" :: "v"(smem) : "memory");

    // copy assignment: thread -> 16B segment; rows 0..63 (+64 in 2nd instr)
    const int rowc = tid >> 2;
    const int segc = (tid & 3) * 8;
    const unsigned short* gA = A  + (size_t)(m0 + rowc) * K + segc;
    const unsigned short* gB = Bt + (size_t)(n0 + rowc) * K + segc;
    const unsigned ldsOff = (unsigned)(rowc * 32 + segc) * 2u;

    auto issue_stage = [&](unsigned buf, int k0) {
        unsigned la = buf * 16384u + ldsOff;           // A tile
        unsigned lb = buf * 16384u + 8192u + ldsOff;   // B tile
        const unsigned short* a0 = gA + k0;
        const unsigned short* a1 = a0 + (size_t)64 * K;
        const unsigned short* b0 = gB + k0;
        const unsigned short* b1 = b0 + (size_t)64 * K;
        asm volatile("global_load_async_to_lds_b128 %0, %1, off"
                     :: "v"(la), "v"(a0), "v"(smem) : "memory");
        asm volatile("global_load_async_to_lds_b128 %0, %1, off"
                     :: "v"(la + 4096u), "v"(a1) : "memory");
        asm volatile("global_load_async_to_lds_b128 %0, %1, off"
                     :: "v"(lb), "v"(b0) : "memory");
        asm volatile("global_load_async_to_lds_b128 %0, %1, off"
                     :: "v"(lb + 4096u), "v"(b1) : "memory");
    };

    auto compute = [&](unsigned buf) {
        const unsigned short* At  = smem + buf * 8192u;          // elem offsets
        const unsigned short* Btl = At + 4096;
        Frag af[2];
        #pragma unroll
        for (int mt = 0; mt < 2; ++mt) {
            const unsigned short* p = At + (wm + mt * 16 + ln) * 32 + g * 8;
            af[mt].q[0] = *(const u32x4*)(p);
            af[mt].q[1] = *(const u32x4*)(p + 16);
        }
        #pragma unroll
        for (int t = 0; t < 4; ++t) {
            const unsigned short* p = Btl + (wn + t * 16 + ln) * 32 + g * 8;
            Frag bf;
            bf.q[0] = *(const u32x4*)(p);
            bf.q[1] = *(const u32x4*)(p + 16);
            acc[0][t] = wmma_bf16(af[0].v, bf.v, acc[0][t]);
            acc[1][t] = wmma_bf16(af[1].v, bf.v, acc[1][t]);
        }
    };

    const int nk = K / 32;
    issue_stage(0, 0);
    unsigned bufR = 0, bufW = 1;
    // steady state: next stage in flight while current buffer is consumed.
    // unroll(disable): keep accumulators in fixed registers (no phi copies).
    #pragma clang loop unroll(disable)
    for (int i = 0; i < nk - 1; ++i) {
        issue_stage(bufW, (i + 1) * 32);
        asm volatile("s_wait_asynccnt 0x4" ::: "memory");  // current stage landed
        __syncthreads();                                   // visible to all waves
        compute(bufR);
        bufR = (bufR == 2u) ? 0u : bufR + 1u;
        bufW = (bufW == 2u) ? 0u : bufW + 1u;
    }
    // epilogue
    asm volatile("s_wait_asynccnt 0x0" ::: "memory");
    __syncthreads();
    compute(bufR);
}

// QKV GEMM + bias + pose bias, scatter to Q[B,H,N,D], K[B,H,N,D], Vt[B,H,D,N]
__global__ __launch_bounds__(256) void qkv_gemm(const unsigned short* __restrict__ xb,
                                                const unsigned short* __restrict__ WqkvT,
                                                const float* __restrict__ bqkv,
                                                const float* __restrict__ pose_emb,
                                                unsigned short* __restrict__ Qws,
                                                unsigned short* __restrict__ Kws,
                                                unsigned short* __restrict__ Vtws) {
    __shared__ __align__(16) unsigned short smem[24576];   // 48 KB, 3 buffers
    const int lane = threadIdx.x & 31, wave = threadIdx.x >> 5;
    const int g = lane >> 4, ln = lane & 15;
    const int wm = (wave & 3) * 32, wn = (wave >> 2) * 64;
    const int m0 = blockIdx.y * 128;
    const int n0 = blockIdx.x * 128;

    v8f acc[2][4] = {};
    gemm_block_128x128(xb, WqkvT, DIM, m0, n0, smem, acc);

    const float qscale = 0.18033688011112042f;    // 0.125 * log2(e)
    #pragma unroll
    for (int mt = 0; mt < 2; ++mt) {
        const int rowbase = m0 + wm + mt * 16;
        const int b = rowbase >> 10;              // 16-row strip stays in-batch
        #pragma unroll
        for (int t = 0; t < 4; ++t) {
            int cidx = n0 + wn + t * 16 + ln;     // output column; w uniform/tile
            int w = cidx >> 10;
            int hd = cidx & 1023;
            int h = hd >> 6, d = hd & 63;
            float bias = bqkv[cidx] + pose_emb[(size_t)b * (3 * DIM) + cidx];
            size_t bh = (size_t)b * HEADS + h;
            #pragma unroll
            for (int r = 0; r < 8; ++r) {
                int row = rowbase + r + 8 * g;
                int n = row & (NSEQ - 1);
                float v = acc[mt][t][r] + bias;
                if (w == 0) {
                    Qws[(bh * NSEQ + n) * HDIM + d] = f32_to_bf16(v * qscale);
                } else if (w == 1) {
                    Kws[(bh * NSEQ + n) * HDIM + d] = f32_to_bf16(v);
                } else {
                    Vtws[(bh * HDIM + d) * NSEQ + n] = f32_to_bf16(v);
                }
            }
        }
    }
}

// ---------------------------------------------------------------------------
// Flash-style attention. One wave handles 16 query rows; 32-key chunks.
// Q already carries the 0.125*log2e scale, so exp2 is the softmax base.
// All K and V fragments for a chunk are issued up-front so V latency hides
// behind the S-WMMAs and the softmax VALU work.
__global__ __launch_bounds__(256) void attn(const unsigned short* __restrict__ Qws,
                                            const unsigned short* __restrict__ Kws,
                                            const unsigned short* __restrict__ Vtws,
                                            unsigned short* __restrict__ Ows) {
    __shared__ __align__(16) unsigned short Plds[8][16 * 32];
    const int lane = threadIdx.x & 31, wave = threadIdx.x >> 5;
    const int g = lane >> 4, ln = lane & 15;
    const int bh = blockIdx.x;                 // 0..127
    const int b = bh >> 4, h = bh & 15;
    const int q0 = blockIdx.y * 128 + wave * 16;

    const unsigned short* Qbh = Qws  + (size_t)bh * NSEQ * HDIM;
    const unsigned short* Kbh = Kws  + (size_t)bh * NSEQ * HDIM;
    const unsigned short* Vbh = Vtws + (size_t)bh * HDIM * NSEQ;

    // Q A-fragments for d-chunks [0,32) and [32,64), live across the loop
    Frag qf0, qf1;
    {
        const unsigned short* qp = Qbh + (size_t)(q0 + ln) * HDIM + g * 8;
        qf0.q[0] = *(const u32x4*)(qp);       qf0.q[1] = *(const u32x4*)(qp + 16);
        qf1.q[0] = *(const u32x4*)(qp + 32);  qf1.q[1] = *(const u32x4*)(qp + 48);
    }

    float rmax[8], rsum[8];
    #pragma unroll
    for (int r = 0; r < 8; ++r) { rmax[r] = -3.0e38f; rsum[r] = 0.0f; }
    v8f O[4] = {};
    unsigned short* Pw = &Plds[wave][0];

    for (int kc = 0; kc < NSEQ; kc += 32) {
        // ---- issue ALL loads for this chunk up-front ----
        const unsigned short* kp0 = Kbh + (size_t)(kc + ln) * HDIM + g * 8;
        const unsigned short* kp1 = kp0 + 16 * HDIM;
        Frag kf00, kf01, kf10, kf11, vf[4];
        kf00.q[0] = *(const u32x4*)(kp0);       kf00.q[1] = *(const u32x4*)(kp0 + 16);
        kf01.q[0] = *(const u32x4*)(kp0 + 32);  kf01.q[1] = *(const u32x4*)(kp0 + 48);
        kf10.q[0] = *(const u32x4*)(kp1);       kf10.q[1] = *(const u32x4*)(kp1 + 16);
        kf11.q[0] = *(const u32x4*)(kp1 + 32);  kf11.q[1] = *(const u32x4*)(kp1 + 48);
        #pragma unroll
        for (int t = 0; t < 4; ++t) {
            const unsigned short* vp = Vbh + (size_t)(t * 16 + ln) * NSEQ + kc + g * 8;
            vf[t].q[0] = *(const u32x4*)(vp);
            vf[t].q[1] = *(const u32x4*)(vp + 16);
        }
        if (kc + 32 < NSEQ) {
            __builtin_prefetch(kp0 + 32 * HDIM, 0, 0);
            __builtin_prefetch(Vbh + (size_t)ln * NSEQ + kc + 32, 0, 0);
        }

        // ---- S = Q K^T (already in log2 domain) ----
        v8f S0 = {}, S1 = {};
        S0 = wmma_bf16(qf0.v, kf00.v, S0);
        S0 = wmma_bf16(qf1.v, kf01.v, S0);
        S1 = wmma_bf16(qf0.v, kf10.v, S1);
        S1 = wmma_bf16(qf1.v, kf11.v, S1);

        // ---- online softmax; rows live across a 16-lane half per VGPR ----
        #pragma unroll
        for (int r = 0; r < 8; ++r) {
            float mx = fmaxf(S0[r], S1[r]);
            mx = fmaxf(mx, __shfl_xor(mx, 1, 16));
            mx = fmaxf(mx, __shfl_xor(mx, 2, 16));
            mx = fmaxf(mx, __shfl_xor(mx, 4, 16));
            mx = fmaxf(mx, __shfl_xor(mx, 8, 16));
            float mnew  = fmaxf(rmax[r], mx);
            float alpha = exp2f(rmax[r] - mnew);
            float p0 = exp2f(S0[r] - mnew);
            float p1 = exp2f(S1[r] - mnew);
            float ls = p0 + p1;
            ls += __shfl_xor(ls, 1, 16);
            ls += __shfl_xor(ls, 2, 16);
            ls += __shfl_xor(ls, 4, 16);
            ls += __shfl_xor(ls, 8, 16);
            rsum[r] = rsum[r] * alpha + ls;
            rmax[r] = mnew;
            #pragma unroll
            for (int t = 0; t < 4; ++t) O[t][r] *= alpha;
            int prow = r + 8 * g;
            Pw[prow * 32 + ln]      = f32_to_bf16(p0);
            Pw[prow * 32 + 16 + ln] = f32_to_bf16(p1);
        }

        // ---- P A-fragment from LDS (wave-private region) ----
        Frag pf;
        {
            const unsigned short* pp = Pw + ln * 32 + g * 8;
            pf.q[0] = *(const u32x4*)(pp);
            pf.q[1] = *(const u32x4*)(pp + 16);
        }
        // ---- O += P(16x32) * V(32x64); V fragments already resident ----
        #pragma unroll
        for (int t = 0; t < 4; ++t) O[t] = wmma_bf16(pf.v, vf[t].v, O[t]);
    }

    // normalize and scatter to merged [B,N,DIM] bf16
    #pragma unroll
    for (int t = 0; t < 4; ++t) {
        #pragma unroll
        for (int r = 0; r < 8; ++r) {
            int n = q0 + r + 8 * g;
            int d = h * HDIM + t * 16 + ln;
            float v = O[t][r] / rsum[r];
            Ows[(size_t)(b * NSEQ + n) * DIM + d] = f32_to_bf16(v);
        }
    }
}

// ---------------------------------------------------------------------------
// output projection: f32 result
__global__ __launch_bounds__(256) void proj_gemm(const unsigned short* __restrict__ Ab,
                                                 const unsigned short* __restrict__ WprojT,
                                                 const float* __restrict__ bias,
                                                 float* __restrict__ out) {
    __shared__ __align__(16) unsigned short smem[24576];   // 48 KB, 3 buffers
    const int lane = threadIdx.x & 31, wave = threadIdx.x >> 5;
    const int g = lane >> 4, ln = lane & 15;
    const int wm = (wave & 3) * 32, wn = (wave >> 2) * 64;
    const int m0 = blockIdx.y * 128;
    const int n0 = blockIdx.x * 128;

    v8f acc[2][4] = {};
    gemm_block_128x128(Ab, WprojT, DIM, m0, n0, smem, acc);

    #pragma unroll
    for (int mt = 0; mt < 2; ++mt) {
        const int rowbase = m0 + wm + mt * 16;
        #pragma unroll
        for (int t = 0; t < 4; ++t) {
            int n = n0 + wn + t * 16 + ln;
            float bv = bias[n];
            #pragma unroll
            for (int r = 0; r < 8; ++r) {
                int row = rowbase + r + 8 * g;
                out[(size_t)row * DIM + n] = acc[mt][t][r] + bv;
            }
        }
    }
}

// ---------------------------------------------------------------------------
extern "C" void kernel_launch(void* const* d_in, const int* in_sizes, int n_in,
                              void* d_out, int out_size, void* d_ws, size_t ws_size,
                              hipStream_t stream) {
    const float* x     = (const float*)d_in[0];
    const float* pose  = (const float*)d_in[1];
    const float* Wqkv  = (const float*)d_in[2];
    const float* bqkv  = (const float*)d_in[3];
    const float* W1    = (const float*)d_in[4];
    const float* b1    = (const float*)d_in[5];
    const float* W2    = (const float*)d_in[6];
    const float* b2    = (const float*)d_in[7];
    const float* Wproj = (const float*)d_in[8];
    const float* bproj = (const float*)d_in[9];
    float* out = (float*)d_out;

    char* p = (char*)d_ws;
    auto alloc = [&](size_t bytes) -> char* {
        char* r = p; p += (bytes + 255) & ~(size_t)255; return r;
    };
    unsigned short* xb      = (unsigned short*)alloc((size_t)MTOT * DIM * 2);
    unsigned short* WqkvT   = (unsigned short*)alloc((size_t)3 * DIM * DIM * 2);
    unsigned short* WprojT  = (unsigned short*)alloc((size_t)DIM * DIM * 2);
    float*          poseEmb = (float*)         alloc((size_t)BSZ * 3 * DIM * 4);
    unsigned short* Qws     = (unsigned short*)alloc((size_t)BSZ * HEADS * NSEQ * HDIM * 2);
    unsigned short* Kws     = (unsigned short*)alloc((size_t)BSZ * HEADS * NSEQ * HDIM * 2);
    unsigned short* Vtws    = (unsigned short*)alloc((size_t)BSZ * HEADS * HDIM * NSEQ * 2);
    unsigned short* Ows     = (unsigned short*)alloc((size_t)MTOT * DIM * 2);

    cvt_f32_bf16<<<(MTOT * DIM) / 256, 256, 0, stream>>>(x, xb, MTOT * DIM);
    transpose_cvt<<<dim3((3 * DIM) / 32, DIM / 32), dim3(32, 8), 0, stream>>>(
        Wqkv, WqkvT, DIM, 3 * DIM);
    transpose_cvt<<<dim3(DIM / 32, DIM / 32), dim3(32, 8), 0, stream>>>(
        Wproj, WprojT, DIM, DIM);
    pose_mlp<<<dim3((3 * DIM) / 256, BSZ), 256, 0, stream>>>(
        pose, W1, b1, W2, b2, poseEmb);
    qkv_gemm<<<dim3((3 * DIM) / 128, MTOT / 128), 256, 0, stream>>>(
        xb, WqkvT, bqkv, poseEmb, Qws, Kws, Vtws);
    attn<<<dim3(BSZ * HEADS, NSEQ / 128), 256, 0, stream>>>(Qws, Kws, Vtws, Ows);
    proj_gemm<<<dim3(DIM / 128, MTOT / 128), 256, 0, stream>>>(Ows, WprojT, bproj, out);
}
